// Baseline_21775484190957
// MI455X (gfx1250) — matrix operand order repaired and checked by reading, not compiled, not run on hardware
//
#include <hip/hip_runtime.h>

// Problem constants (match reference)
#define NNODES 100000
#define NEDGES 1600000
#define DIN    48
#define HID    256
#define HHALF  128
#define NOUTC  40

typedef __attribute__((ext_vector_type(16))) __bf16       bf16x16;
typedef __attribute__((ext_vector_type(8)))  float        floatx8;
typedef __attribute__((ext_vector_type(4)))  unsigned int uintx4;

union Frag16 {
    bf16x16 v;
    uintx4  u[2];
};

__device__ __forceinline__ unsigned short f32_to_bf16(float f) {
    unsigned int u = __float_as_uint(f);
    u += 0x7FFFu + ((u >> 16) & 1u);   // round-to-nearest-even
    return (unsigned short)(u >> 16);
}

__device__ __forceinline__ floatx8 wmma_bf16(const Frag16& a, const Frag16& b, floatx8 c) {
    return __builtin_amdgcn_wmma_f32_16x16x32_bf16(false, a.v, false, b.v,
                                                   (short)0, c, false, false);
}

// ---------------------------------------------------------------------------
// Scatter-add: c[dst[e], f] += x[src[e], f]   (segment_sum)
// One lane per (edge, feature); Dpad = 2^shift lanes per edge for alignment.
// ---------------------------------------------------------------------------
__global__ __launch_bounds__(256) void scatter_add_kernel(
    const float* __restrict__ x, const int* __restrict__ src,
    const int* __restrict__ dst, float* __restrict__ c,
    long long nE, int D, int shift)
{
    long long tid = (long long)blockIdx.x * 256 + threadIdx.x;
    long long e  = tid >> shift;
    int       f  = (int)(tid & ((1 << shift) - 1));
    if (e < nE && f < D) {
        int s = src[e];
        int d = dst[e];
        atomicAdd(&c[(long long)d * D + f], x[(long long)s * D + f]);
    }
}

// ---------------------------------------------------------------------------
// Weight convert: fp32 W[K][Nout] -> bf16 Wt[NoutPad][KPAD] (transposed, padded)
// ---------------------------------------------------------------------------
__global__ __launch_bounds__(256) void wconv_kernel(
    const float* __restrict__ W, unsigned short* __restrict__ Wt,
    int K, int Nout, int KPAD, int NoutPad)
{
    int idx = blockIdx.x * 256 + threadIdx.x;
    int total = NoutPad * KPAD;
    if (idx >= total) return;
    int n = idx / KPAD;
    int k = idx - n * KPAD;
    float v = (n < Nout && k < K) ? W[(long long)k * Nout + n] : 0.0f;
    Wt[idx] = f32_to_bf16(v);
}

// ---------------------------------------------------------------------------
// Fused 2-layer MLP:  out = act( relu( concat(in0,in1,in2) @ Wa + ba ) @ Wb + bb )
// 256 threads (8 waves), 32 rows per block; hidden [32,256] lives in LDS only.
// Wat: bf16 [HID][KPAD] (row n = hidden unit n), Wbt: bf16 [HOUTPAD][HID].
// ---------------------------------------------------------------------------
template<int KPAD, int HOUTPAD, bool RELU>
__global__ __launch_bounds__(256) void fused_mlp_kernel(
    const float* __restrict__ in0, int K0,
    const float* __restrict__ in1, int K1,
    const float* __restrict__ in2, int K2,
    const unsigned short* __restrict__ Wat, const float* __restrict__ ba,
    const unsigned short* __restrict__ Wbt, const float* __restrict__ bb,
    float* __restrict__ out, int Hout, int n_rows)
{
    __shared__ __align__(16) unsigned short Xtile[32 * KPAD];
    __shared__ __align__(16) unsigned short Htile[32 * HID];

    const int rowBase = blockIdx.x * 32;
    const int tid = threadIdx.x;

    // Stage 1: build bf16 input tile (concat + zero pad)
    for (int idx = tid; idx < 32 * KPAD; idx += 256) {
        int r = idx / KPAD;
        int c = idx - r * KPAD;
        int gr = rowBase + r;
        float v = 0.0f;
        if (gr < n_rows) {
            if (c < K0)                v = in0[(long long)gr * K0 + c];
            else if (c < K0 + K1)      v = in1[(long long)gr * K1 + (c - K0)];
            else if (c < K0 + K1 + K2) v = in2[(long long)gr * K2 + (c - K0 - K1)];
        }
        Xtile[idx] = f32_to_bf16(v);
    }
    __syncthreads();

    const int wave = tid >> 5;
    const int lane = tid & 31;
    const int mrow = lane & 15;   // row (A) or column (B/C) within 16-tile
    const int hi   = lane >> 4;   // lane-half selector per CDNA5 WMMA layout

    // Stage 2: layer A — each wave computes 4 of the 2x16 output tiles
    {
        const int rt     = wave >> 2;        // 0..1  row tile
        const int ctBase = (wave & 3) * 4;   // 0,4,8,12
        floatx8 acc[4] = {};
        const unsigned short* xrow = &Xtile[(rt * 16 + mrow) * KPAD];
        #pragma unroll
        for (int ks = 0; ks < KPAD / 32; ++ks) {
            const int k0 = ks * 32;
            Frag16 a;   // A 16x32 bf16: two 8-half runs per lane
            a.u[0] = *(const uintx4*)(xrow + k0 + 8 * hi);
            a.u[1] = *(const uintx4*)(xrow + k0 + 16 + 8 * hi);
            #pragma unroll
            for (int j = 0; j < 4; ++j) {
                const int nc = (ctBase + j) * 16 + mrow;
                const unsigned short* wrow = Wat + (long long)nc * KPAD + k0 + 8 * hi;
                Frag16 b;   // B 32x16 bf16: two 8-half runs per lane
                b.u[0] = *(const uintx4*)(wrow);
                b.u[1] = *(const uintx4*)(wrow + 16);
                acc[j] = wmma_bf16(a, b, acc[j]);
            }
        }
        #pragma unroll
        for (int j = 0; j < 4; ++j) {
            const int nc = (ctBase + j) * 16 + mrow;
            const float bias = ba[nc];
            #pragma unroll
            for (int e = 0; e < 8; ++e) {
                const int m = rt * 16 + e + 8 * hi;   // C/D layout: M = e + 8*hi
                float v = acc[j][e] + bias;
                v = v > 0.0f ? v : 0.0f;              // inner MLP ReLU
                Htile[m * HID + nc] = f32_to_bf16(v);
            }
        }
    }
    __syncthreads();

    // Stage 3: layer B — 2 x (HOUTPAD/16) tiles split over 8 waves
    const int nct = HOUTPAD / 16;
    for (int t = wave; t < 2 * nct; t += 8) {
        const int rt = t & 1;
        const int ct = t >> 1;
        floatx8 acc = {};
        const unsigned short* hrow  = &Htile[(rt * 16 + mrow) * HID];
        const unsigned short* wrow0 = Wbt + (long long)(ct * 16 + mrow) * HID;
        #pragma unroll
        for (int ks = 0; ks < HID / 32; ++ks) {
            const int k0 = ks * 32;
            Frag16 a;
            a.u[0] = *(const uintx4*)(hrow + k0 + 8 * hi);
            a.u[1] = *(const uintx4*)(hrow + k0 + 16 + 8 * hi);
            Frag16 b;
            b.u[0] = *(const uintx4*)(wrow0 + k0 + 8 * hi);
            b.u[1] = *(const uintx4*)(wrow0 + k0 + 16 + 8 * hi);
            acc = wmma_bf16(a, b, acc);
        }
        const int nc = ct * 16 + mrow;
        if (nc < Hout) {
            const float bias = bb[nc];
            #pragma unroll
            for (int e = 0; e < 8; ++e) {
                const int gr = rowBase + rt * 16 + e + 8 * hi;
                if (gr < n_rows) {
                    float v = acc[e] + bias;
                    if (RELU) v = v > 0.0f ? v : 0.0f;   // outer ReLU (h1/h2)
                    out[(long long)gr * Hout + nc] = v;
                }
            }
        }
    }
}

// ---------------------------------------------------------------------------
// Row-wise log-softmax over C=40 columns
// ---------------------------------------------------------------------------
__global__ __launch_bounds__(256) void logsoftmax_kernel(
    const float* __restrict__ logits, float* __restrict__ out, int n, int C)
{
    int i = blockIdx.x * 256 + threadIdx.x;
    if (i >= n) return;
    const float* row = logits + (long long)i * C;
    float m = -3.402823466e38f;
    for (int j = 0; j < C; ++j) m = fmaxf(m, row[j]);
    float s = 0.0f;
    for (int j = 0; j < C; ++j) s += __expf(row[j] - m);
    float ls = __logf(s);
    float* orow = out + (long long)i * C;
    for (int j = 0; j < C; ++j) orow[j] = row[j] - m - ls;
}

// ---------------------------------------------------------------------------
extern "C" void kernel_launch(void* const* d_in, const int* in_sizes, int n_in,
                              void* d_out, int out_size, void* d_ws, size_t ws_size,
                              hipStream_t stream)
{
    const float* x   = (const float*)d_in[0];
    const int*   ei  = (const int*)d_in[1];
    const int*   src = ei;            // edge_index[0]
    const int*   dst = ei + NEDGES;   // edge_index[1]
    const float* W1a = (const float*)d_in[2];  const float* b1a = (const float*)d_in[3];
    const float* W1b = (const float*)d_in[4];  const float* b1b = (const float*)d_in[5];
    const float* W2a = (const float*)d_in[6];  const float* b2a = (const float*)d_in[7];
    const float* W2b = (const float*)d_in[8];  const float* b2b = (const float*)d_in[9];
    const float* W3a = (const float*)d_in[10]; const float* b3a = (const float*)d_in[11];
    const float* W3b = (const float*)d_in[12]; const float* b3b = (const float*)d_in[13];

    // Workspace carving (256B aligned)
    char*  ws  = (char*)d_ws;
    size_t off = 0;
    auto carve = [&](size_t bytes) -> void* {
        void* p = ws + off;
        off = (off + bytes + 255) & ~(size_t)255;
        return p;
    };
    float* c1     = (float*)carve((size_t)NNODES * DIN   * 4);
    float* h1     = (float*)carve((size_t)NNODES * HHALF * 4);
    float* c2     = (float*)carve((size_t)NNODES * HHALF * 4);
    float* h2     = (float*)carve((size_t)NNODES * HHALF * 4);
    float* c3     = (float*)carve((size_t)NNODES * HHALF * 4);
    float* logits = (float*)carve((size_t)NNODES * NOUTC * 4);
    unsigned short* w1at = (unsigned short*)carve((size_t)HID * 96  * 2);
    unsigned short* w1bt = (unsigned short*)carve((size_t)HHALF * HID * 2);
    unsigned short* w2at = (unsigned short*)carve((size_t)HID * 320 * 2);
    unsigned short* w2bt = (unsigned short*)carve((size_t)HHALF * HID * 2);
    unsigned short* w3at = (unsigned short*)carve((size_t)HID * 320 * 2);
    unsigned short* w3bt = (unsigned short*)carve((size_t)48 * HID * 2);

    // Zero the segment-sum accumulators
    hipMemsetAsync(c1, 0, (size_t)NNODES * DIN   * 4, stream);
    hipMemsetAsync(c2, 0, (size_t)NNODES * HHALF * 4, stream);
    hipMemsetAsync(c3, 0, (size_t)NNODES * HHALF * 4, stream);

    // Weight transpose + bf16 convert (tiny)
    auto wcblocks = [](int n) { return (n + 255) / 256; };
    wconv_kernel<<<wcblocks(HID * 96),    256, 0, stream>>>(W1a, w1at,  96, HID,  96, HID);
    wconv_kernel<<<wcblocks(HHALF * HID), 256, 0, stream>>>(W1b, w1bt, HID, HHALF, HID, HHALF);
    wconv_kernel<<<wcblocks(HID * 320),   256, 0, stream>>>(W2a, w2at, 304, HID, 320, HID);
    wconv_kernel<<<wcblocks(HHALF * HID), 256, 0, stream>>>(W2b, w2bt, HID, HHALF, HID, HHALF);
    wconv_kernel<<<wcblocks(HID * 320),   256, 0, stream>>>(W3a, w3at, 304, HID, 320, HID);
    wconv_kernel<<<wcblocks(48 * HID),    256, 0, stream>>>(W3b, w3bt, HID, NOUTC, HID, 48);

    const int mlpBlocks = (NNODES + 31) / 32;   // 3125

    // Layer 1: c1 = lgconv(x);  h1 = relu(mlp1([c1, x]))
    {
        long long tot = (long long)NEDGES << 6;   // Dpad=64 for D=48
        scatter_add_kernel<<<(unsigned)((tot + 255) / 256), 256, 0, stream>>>(
            x, src, dst, c1, NEDGES, DIN, 6);
        fused_mlp_kernel<96, 128, true><<<mlpBlocks, 256, 0, stream>>>(
            c1, DIN, x, DIN, nullptr, 0, w1at, b1a, w1bt, b1b, h1, HHALF, NNODES);
    }
    // Layer 2: c2 = lgconv(h1);  h2 = relu(mlp2([c2, h1, x]))
    {
        long long tot = (long long)NEDGES << 7;   // Dpad=128 for D=128
        scatter_add_kernel<<<(unsigned)((tot + 255) / 256), 256, 0, stream>>>(
            h1, src, dst, c2, NEDGES, HHALF, 7);
        fused_mlp_kernel<320, 128, true><<<mlpBlocks, 256, 0, stream>>>(
            c2, HHALF, h1, HHALF, x, DIN, w2at, b2a, w2bt, b2b, h2, HHALF, NNODES);
    }
    // Layer 3: c3 = lgconv(h2);  logits = mlp3([c3, h2, x]); log_softmax
    {
        long long tot = (long long)NEDGES << 7;
        scatter_add_kernel<<<(unsigned)((tot + 255) / 256), 256, 0, stream>>>(
            h2, src, dst, c3, NEDGES, HHALF, 7);
        fused_mlp_kernel<320, 48, false><<<mlpBlocks, 256, 0, stream>>>(
            c3, HHALF, h2, HHALF, x, DIN, w3at, b3a, w3bt, b3b, logits, NOUTC, NNODES);
        logsoftmax_kernel<<<(NNODES + 255) / 256, 256, 0, stream>>>(
            logits, (float*)d_out, NNODES, NOUTC);
    }
}
